// DGNN_41583873360060
// MI455X (gfx1250) — matrix operand order repaired and checked by reading, not compiled, not run on hardware
//
#include <hip/hip_runtime.h>
#include <hip/hip_bf16.h>
#include <math.h>

// ---- MI455X / gfx1250 WMMA types ----
typedef __attribute__((ext_vector_type(16))) __bf16 v16bf;
typedef __attribute__((ext_vector_type(8)))  __bf16 v8bf;
typedef __attribute__((ext_vector_type(8)))  float  v8f;
typedef __attribute__((ext_vector_type(4)))  float  v4f;
typedef __attribute__((ext_vector_type(4)))  int    v4i;

#define NN     8192              // graph nodes (M and K of the big GEMM)
#define BB     4                 // batch
#define FF     32                // features
#define COLS   (BB * FF)         // 128 columns of the big GEMM
#define KSPLIT 4                 // K-dimension split for occupancy
#define KCHUNK (NN / KSPLIT)     // 2048 K per wave

// ---------------------------------------------------------------------------
// Kernel 1: repack x[b,j,f] (f32) -> Xt[col=b*32+f][j] (bf16, column-major so
// each WMMA B-fragment lane load is a contiguous 16B run of K), and zero the
// f32 partial-sum accumulator (needed every call: spmm accumulates atomically).
// ---------------------------------------------------------------------------
__global__ __launch_bounds__(256) void dgnn_prep(const float* __restrict__ x,
                                                 __bf16* __restrict__ Xt,
                                                 float* __restrict__ acc) {
    int tid = blockIdx.x * 256 + threadIdx.x;     // 0 .. NN*COLS-1 (1M)
    int col = tid >> 13;                          // / NN
    int j   = tid & (NN - 1);
    int b   = col >> 5;
    int f   = col & 31;
    Xt[tid]  = (__bf16)x[((size_t)b * NN + j) * FF + f];
    acc[tid] = 0.0f;
}

// ---------------------------------------------------------------------------
// Kernel 2: masked SpMM  acc[i, col] += sum_k A[i,k] * Xt[col, k]
// A[i,k] = (time_delay[i,k]/5 <= t && adj[i,k] > 0) ? adj[i,k] : 0, as bf16.
// One wave = 16-row tile x 128 cols x 2048 K; 8 f32 accumulator fragments.
// ---------------------------------------------------------------------------
__device__ __forceinline__ __bf16 mask_cvt(float a, int tdv, int thr) {
    // t >= td//5  <=>  td <= 5t+4  (td >= 0, integer floor division)
    float v = ((tdv <= thr) && (a > 0.0f)) ? a : 0.0f;
    return (__bf16)v;
}

__global__ __launch_bounds__(128) void dgnn_spmm(const float* __restrict__ adj,
                                                 const int*   __restrict__ td,
                                                 const int*   __restrict__ tptr,
                                                 const __bf16* __restrict__ Xt,
                                                 float* __restrict__ acc) {
    const int lane = threadIdx.x & 31;
    const int wave = threadIdx.x >> 5;
    const int lh   = lane >> 4;        // which 16-lane half
    const int nl   = lane & 15;
    const int i0   = (blockIdx.x * 4 + wave) * 16;   // row tile base
    const int row  = i0 + nl;                        // A fragment: M = lane%16
    const int kbeg = blockIdx.y * KCHUNK;
    const int thr  = 5 * (*tptr) + 4;

    const float* __restrict__ arow = adj + (size_t)row * NN;
    const int*   __restrict__ trow = td  + (size_t)row * NN;

    v8f c[8] = {};   // 16x16 f32 accumulators for the 8 column blocks

    #pragma unroll 2
    for (int kb = kbeg; kb < kbeg + KCHUNK; kb += 32) {
        // ---- A fragment (16x32 bf16): ISA layout -> per lane two 8-elem
        // contiguous K runs: [kb + lh*8, +8) and [kb + 16 + lh*8, +8).
        const int k0 = kb + lh * 8;
        v4f a0 = *(const v4f*)(arow + k0);
        v4f a1 = *(const v4f*)(arow + k0 + 4);
        v4f a2 = *(const v4f*)(arow + k0 + 16);
        v4f a3 = *(const v4f*)(arow + k0 + 20);
        v4i t0 = *(const v4i*)(trow + k0);
        v4i t1 = *(const v4i*)(trow + k0 + 4);
        v4i t2 = *(const v4i*)(trow + k0 + 16);
        v4i t3 = *(const v4i*)(trow + k0 + 20);

        // stream-prefetch ahead on the two 256MB operands (global_prefetch_b8)
        __builtin_prefetch(arow + kb + 1024, 0, 0);
        __builtin_prefetch(trow + kb + 1024, 0, 0);

        v16bf afrag;
        #pragma unroll
        for (int e = 0; e < 4; ++e) {
            afrag[e]      = mask_cvt(a0[e], t0[e], thr);
            afrag[4 + e]  = mask_cvt(a1[e], t1[e], thr);
            afrag[8 + e]  = mask_cvt(a2[e], t2[e], thr);
            afrag[12 + e] = mask_cvt(a3[e], t3[e], thr);
        }

        // ---- 8 column blocks: B fragment (32x16 bf16) = per lane two 8-elem
        // K runs of column (cb*16 + nl): [kb + lh*8, +8) and [kb + 16 + lh*8, +8).
        #pragma unroll
        for (int cb = 0; cb < 8; ++cb) {
            const __bf16* xp = Xt + (size_t)(cb * 16 + nl) * NN + kb + lh * 8;
            v8bf blo = *(const v8bf*)(xp);
            v8bf bhi = *(const v8bf*)(xp + 16);
            v16bf bfrag;
            #pragma unroll
            for (int e = 0; e < 8; ++e) { bfrag[e] = blo[e]; bfrag[8 + e] = bhi[e]; }

            c[cb] = __builtin_amdgcn_wmma_f32_16x16x32_bf16(
                        false, afrag, false, bfrag,
                        (short)0, c[cb], false, false);
        }
    }

    // ---- combine K-split partials: C layout -> VGPR r, lane L holds
    // (M = r + 8*(L>>4), N = L&15) of the 16x16 block.
    float* obase = acc + (size_t)i0 * COLS + nl;
    #pragma unroll
    for (int cb = 0; cb < 8; ++cb) {
        #pragma unroll
        for (int r = 0; r < 8; ++r) {
            atomicAdd(obase + (size_t)(r + 8 * lh) * COLS + cb * 16, c[cb][r]);
        }
    }
}

// ---------------------------------------------------------------------------
// Kernel 3: per (b, i) row: h = relu(v @ W1 + b1); out = sigmoid(h @ W2 + b2)
// ---------------------------------------------------------------------------
__global__ __launch_bounds__(256) void dgnn_mlp(const float* __restrict__ acc,
                                                const float* __restrict__ W1,
                                                const float* __restrict__ b1,
                                                const float* __restrict__ W2,
                                                const float* __restrict__ b2,
                                                float* __restrict__ out) {
    __shared__ float sW1[FF * FF];
    __shared__ float sW2[FF * FF];
    __shared__ float sb[2 * FF];
    for (int i = threadIdx.x; i < FF * FF; i += 256) { sW1[i] = W1[i]; sW2[i] = W2[i]; }
    if (threadIdx.x < FF) {
        sb[threadIdx.x]      = b1[threadIdx.x];
        sb[FF + threadIdx.x] = b2[threadIdx.x];
    }
    __syncthreads();

    int idx = blockIdx.x * 256 + threadIdx.x;     // 0 .. BB*NN-1
    int b   = idx >> 13;                          // / NN
    int i   = idx & (NN - 1);

    const float* vptr = acc + (size_t)i * COLS + b * FF;
    float v[FF], h[FF];
    #pragma unroll
    for (int g = 0; g < FF; ++g) v[g] = vptr[g];

    #pragma unroll 4
    for (int f = 0; f < FF; ++f) {
        float s = sb[f];
        #pragma unroll
        for (int g = 0; g < FF; ++g) s += v[g] * sW1[g * FF + f];
        h[f] = fmaxf(s, 0.0f);
    }

    float* op = out + (size_t)idx * FF;           // [b][i][f] layout
    #pragma unroll 4
    for (int f = 0; f < FF; ++f) {
        float s = sb[FF + f];
        #pragma unroll
        for (int g = 0; g < FF; ++g) s += h[g] * sW2[g * FF + f];
        op[f] = 1.0f / (1.0f + __expf(-s));
    }
}

// ---------------------------------------------------------------------------
extern "C" void kernel_launch(void* const* d_in, const int* in_sizes, int n_in,
                              void* d_out, int out_size, void* d_ws, size_t ws_size,
                              hipStream_t stream) {
    const float* x   = (const float*)d_in[0];   // [4, 8192, 32]
    const float* adj = (const float*)d_in[1];   // [8192, 8192]
    const int*   td  = (const int*)  d_in[2];   // [8192, 8192]
    const int*   tp  = (const int*)  d_in[3];   // scalar t
    const float* W1  = (const float*)d_in[4];
    const float* b1  = (const float*)d_in[5];
    const float* W2  = (const float*)d_in[6];
    const float* b2  = (const float*)d_in[7];
    float* out = (float*)d_out;

    // workspace: [0, 2MB) bf16 Xt;  [2MB, 6MB) f32 accumulator
    __bf16* Xt = (__bf16*)d_ws;
    float*  acc = (float*)((char*)d_ws + (size_t)COLS * NN * sizeof(__bf16));

    dgnn_prep<<<(NN * COLS) / 256, 256, 0, stream>>>(x, Xt, acc);

    dim3 grid(NN / 16 / 4, KSPLIT);               // 128 x 4 blocks, 4 waves each
    dgnn_spmm<<<grid, 128, 0, stream>>>(adj, td, tp, Xt, acc);

    dgnn_mlp<<<(BB * NN) / 256, 256, 0, stream>>>(acc, W1, b1, W2, b2, out);
}
